// TransformerEncoder_15693810500179
// MI455X (gfx1250) — matrix operand order repaired and checked by reading, not compiled, not run on hardware
//
#include <hip/hip_runtime.h>
#include <hip/hip_bf16.h>

#define NNODES 25000
#define NEDGES 400000
#define DRAW   127
#define HCH    256
#define NHEAD  8
#define NCOLS  1024   // q|k|v|s concatenated: 4*256

typedef __attribute__((ext_vector_type(16))) __bf16 v16bf;
typedef __attribute__((ext_vector_type(8)))  float  v8f;

union ABu { v16bf v; unsigned u[8]; };

__device__ __forceinline__ unsigned short f2bf(float f) {
  unsigned u = __float_as_uint(f);
  unsigned r = u + 0x7FFFu + ((u >> 16) & 1u);   // round-to-nearest-even
  return (unsigned short)(r >> 16);
}

// order-preserving float<->uint encoding for atomic segment-max
__device__ __forceinline__ unsigned encf(float f) {
  unsigned u = __float_as_uint(f);
  return (u & 0x80000000u) ? ~u : (u | 0x80000000u);
}
__device__ __forceinline__ float decf(unsigned e) {
  unsigned u = (e & 0x80000000u) ? (e & 0x7FFFFFFFu) : ~e;
  return __uint_as_float(u);
}

__global__ void zero_u32_kernel(unsigned* __restrict__ p, int n) {
  int i = blockIdx.x * blockDim.x + threadIdx.x;
  if (i < n) p[i] = 0u;
}

// Build layer-0 input: [N,128] = concat(where(mask, fill, x_orig), mask) -> bf16
__global__ void prep_kernel(const float* __restrict__ x_orig,
                            const float* __restrict__ mask,
                            const float* __restrict__ fill,
                            unsigned short* __restrict__ xb) {
  int i = blockIdx.x * blockDim.x + threadIdx.x;
  if (i >= NNODES * 128) return;
  int row = i >> 7, col = i & 127;
  float m = mask[row];
  float v = (col < DRAW) ? ((m > 0.f) ? fill[col] : x_orig[row * DRAW + col]) : m;
  xb[i] = f2bf(v);
}

// Fuse Wq|Wk|Wv|Ws -> bf16 [1024, Kdim] and biases -> [1024]
__global__ void pack_w_kernel(const float* __restrict__ Wq, const float* __restrict__ Wk,
                              const float* __restrict__ Wv, const float* __restrict__ Ws,
                              const float* __restrict__ bq, const float* __restrict__ bk,
                              const float* __restrict__ bv, const float* __restrict__ bs,
                              unsigned short* __restrict__ wallb,
                              float* __restrict__ ball, int Kdim) {
  int i = blockIdx.x * blockDim.x + threadIdx.x;
  if (i >= NCOLS * Kdim) return;
  int nc = i / Kdim, k = i - nc * Kdim;
  int mat = nc >> 8, o = nc & 255;
  const float* W = (mat == 0) ? Wq : (mat == 1) ? Wk : (mat == 2) ? Wv : Ws;
  wallb[i] = f2bf(W[o * Kdim + k]);
  if (k == 0) {
    const float* B = (mat == 0) ? bq : (mat == 1) ? bk : (mat == 2) ? bv : bs;
    ball[nc] = B[o];
  }
}

// Y[N,1024] = Xb[N,K] * Wall^T + ball, via V_WMMA_F32_16X16X32_BF16.
// One 16x64 output tile per wave (4 accumulators): A fragment loaded once per
// K-step, reused across 4 B fragments. K is a template constant and the K-loop
// fully unrolled so the scheduler can hoist next-step fragment loads above the
// current step's WMMAs (hide L1/L2 latency behind the matrix pipe) instead of
// recycling one B register set with s_wait_loadcnt 0 before every WMMA.
// Fragment loads use contiguous K (k = 16*hi + s) for A (16x32) and B (32x16) alike
// (2 consecutive K bf16 per dword -> compiler coalesces into b128 clauses).
template <int KD>
__global__ void gemm_wmma_kernel(const unsigned short* __restrict__ xb,
                                 const unsigned short* __restrict__ wallb,
                                 const float* __restrict__ ball,
                                 float* __restrict__ yall) {
  const int tilesQ = NCOLS / 64;                  // 16 column-groups of 64
  const int tilesM = (NNODES + 15) / 16;          // 1563
  int wave = (int)((blockIdx.x * blockDim.x + threadIdx.x) >> 5);
  if (wave >= tilesM * tilesQ) return;            // wave-uniform: EXEC all-ones inside
  int tm = wave / tilesQ, tq = wave - tm * tilesQ;
  int lane = threadIdx.x & 31;
  int l15 = lane & 15, hi = lane >> 4;

  int mA = tm * 16 + l15;
  if (mA >= NNODES) mA = NNODES - 1;              // clamp; stores are guarded
  const unsigned short* arow = xb + (size_t)mA * KD + (hi << 4);       // A: +16 K for hi half

  int nc0 = tq * 64 + l15;
  const unsigned short* brow0 = wallb + (size_t)(nc0 +  0) * KD + (hi << 4); // B: +16 K hi half
  const unsigned short* brow1 = wallb + (size_t)(nc0 + 16) * KD + (hi << 4);
  const unsigned short* brow2 = wallb + (size_t)(nc0 + 32) * KD + (hi << 4);
  const unsigned short* brow3 = wallb + (size_t)(nc0 + 48) * KD + (hi << 4);

  v8f acc0 = {0.f,0.f,0.f,0.f,0.f,0.f,0.f,0.f};
  v8f acc1 = acc0, acc2 = acc0, acc3 = acc0;

#pragma unroll
  for (int k0 = 0; k0 < KD; k0 += 32) {
    ABu a, b0, b1, b2, b3;
#pragma unroll
    for (int r = 0; r < 8; ++r) {
      int ka = k0 + 2 * r;                                   // A VGPR r -> K pair
      a.u[r] = *(const unsigned*)(arow + ka);
      int kb = k0 + 2 * r;                                   // B VGPR r -> K pair
      b0.u[r] = *(const unsigned*)(brow0 + kb);
      b1.u[r] = *(const unsigned*)(brow1 + kb);
      b2.u[r] = *(const unsigned*)(brow2 + kb);
      b3.u[r] = *(const unsigned*)(brow3 + kb);
    }
    acc0 = __builtin_amdgcn_wmma_f32_16x16x32_bf16(false, a.v, false, b0.v,
                                                   (short)0, acc0, false, false);
    acc1 = __builtin_amdgcn_wmma_f32_16x16x32_bf16(false, a.v, false, b1.v,
                                                   (short)0, acc1, false, false);
    acc2 = __builtin_amdgcn_wmma_f32_16x16x32_bf16(false, a.v, false, b2.v,
                                                   (short)0, acc2, false, false);
    acc3 = __builtin_amdgcn_wmma_f32_16x16x32_bf16(false, a.v, false, b3.v,
                                                   (short)0, acc3, false, false);
  }

  int mbase = tm * 16 + (hi << 3);                 // C/D: VGPR i -> row i (+8 for hi half)
#pragma unroll
  for (int j = 0; j < 4; ++j) {
    int col = nc0 + j * 16;
    float bias = ball[col];
    const v8f& acc = (j == 0) ? acc0 : (j == 1) ? acc1 : (j == 2) ? acc2 : acc3;
#pragma unroll
    for (int i = 0; i < 8; ++i) {
      int m = mbase + i;
      if (m < NNODES) yall[(size_t)m * NCOLS + col] = acc[i] + bias;
    }
  }
}

// alpha[e,h] = <q[dst,h,:], k[src,h,:]> / sqrt(32); segment-max via encoded atomicMax
__global__ void attn_logits_kernel(const long long* __restrict__ ei,
                                   const float* __restrict__ yall,
                                   float* __restrict__ alpha,
                                   unsigned* __restrict__ amaxE) {
  int idx = blockIdx.x * blockDim.x + threadIdx.x;
  if (idx >= NEDGES * NHEAD) return;
  int e = idx >> 3, h = idx & 7;
  int s = (int)ei[e];
  int d = (int)ei[NEDGES + e];
  const float4* q4 = (const float4*)(yall + (size_t)d * NCOLS + h * 32);
  const float4* k4 = (const float4*)(yall + (size_t)s * NCOLS + 256 + h * 32);
  float acc = 0.f;
#pragma unroll
  for (int i = 0; i < 8; ++i) {
    float4 qv = q4[i], kv = k4[i];
    acc += qv.x * kv.x + qv.y * kv.y + qv.z * kv.z + qv.w * kv.w;
  }
  float a = acc * 0.17677669529663687f;   // 1/sqrt(32)
  alpha[idx] = a;
  atomicMax(&amaxE[d * NHEAD + h], encf(a));
}

// ex = exp(alpha - amax[dst]); denom[dst] += ex; agg[dst] += ex * v[src]
__global__ void attn_accum_kernel(const long long* __restrict__ ei,
                                  const float* __restrict__ yall,
                                  const float* __restrict__ alpha,
                                  const unsigned* __restrict__ amaxE,
                                  float* __restrict__ denom,
                                  float* __restrict__ agg) {
  int idx = blockIdx.x * blockDim.x + threadIdx.x;
  if (idx >= NEDGES * NHEAD) return;
  int e = idx >> 3, h = idx & 7;
  int s = (int)ei[e];
  int d = (int)ei[NEDGES + e];
  float ex = __expf(alpha[idx] - decf(amaxE[d * NHEAD + h]));
  atomicAdd(&denom[d * NHEAD + h], ex);
  const float4* v4 = (const float4*)(yall + (size_t)s * NCOLS + 512 + h * 32);
  float* op = agg + (size_t)d * HCH + h * 32;
#pragma unroll
  for (int i = 0; i < 8; ++i) {
    float4 vv = v4[i];
    atomicAdd(op + 4 * i + 0, ex * vv.x);
    atomicAdd(op + 4 * i + 1, ex * vv.y);
    atomicAdd(op + 4 * i + 2, ex * vv.z);
    atomicAdd(op + 4 * i + 3, ex * vv.w);
  }
}

// Per-node: softmax normalize, beta gate, blend, LayerNorm, ReLU/residual, requantize
__global__ void finish_kernel(const float* __restrict__ yall,
                              const float* __restrict__ agg,
                              const float* __restrict__ denom,
                              const float* __restrict__ Wb,
                              const float* __restrict__ lng,
                              const float* __restrict__ lnb,
                              const float* __restrict__ xres,      // residual or null
                              float* __restrict__ xout,
                              unsigned short* __restrict__ xbout,  // bf16 out or null
                              int do_relu) {
  __shared__ float red[256];
  int n = blockIdx.x, t = threadIdx.x, h = t >> 5;
  float den = denom[n * NHEAD + h];
  float o = agg[(size_t)n * HCH + t];
  o = (den > 0.f) ? (o / den) : 0.f;                 // empty segment -> 0
  float xr = yall[(size_t)n * NCOLS + 768 + t];

  // beta = sigmoid(Wb . [out, xr, out-xr])
  red[t] = Wb[t] * o + Wb[256 + t] * xr + Wb[512 + t] * (o - xr);
  __syncthreads();
  for (int st = 128; st > 0; st >>= 1) { if (t < st) red[t] += red[t + st]; __syncthreads(); }
  float beta = 1.f / (1.f + __expf(-red[0]));
  __syncthreads();

  float y = beta * xr + (1.f - beta) * o;

  red[t] = y; __syncthreads();
  for (int st = 128; st > 0; st >>= 1) { if (t < st) red[t] += red[t + st]; __syncthreads(); }
  float mu = red[0] * (1.f / 256.f);
  __syncthreads();
  float dy = y - mu;
  red[t] = dy * dy; __syncthreads();
  for (int st = 128; st > 0; st >>= 1) { if (t < st) red[t] += red[t + st]; __syncthreads(); }
  float var = red[0] * (1.f / 256.f);
  __syncthreads();

  float yn = dy * rsqrtf(var + 1e-5f) * lng[t] + lnb[t];
  if (do_relu) yn = fmaxf(yn, 0.f);
  if (xres) yn += xres[(size_t)n * HCH + t];
  xout[(size_t)n * HCH + t] = yn;
  if (xbout) xbout[(size_t)n * HCH + t] = f2bf(yn);
}

extern "C" void kernel_launch(void* const* d_in, const int* in_sizes, int n_in,
                              void* d_out, int out_size, void* d_ws, size_t ws_size,
                              hipStream_t stream) {
  (void)in_sizes; (void)n_in; (void)out_size; (void)ws_size;
  const float*     x_orig = (const float*)d_in[0];
  const long long* ei     = (const long long*)d_in[1];   // int64 [2,E]
  const float*     mask   = (const float*)d_in[2];
  const float*     fill   = (const float*)d_in[3];

  char* ws = (char*)d_ws;
  size_t off = 0;
  auto carve = [&](size_t bytes) -> void* {
    void* p = ws + off; off += (bytes + 255) & ~(size_t)255; return p;
  };
  unsigned short* xb    = (unsigned short*)carve((size_t)NNODES * HCH * 2);
  float*          xfA   = (float*)carve((size_t)NNODES * HCH * 4);
  float*          xfB   = (float*)carve((size_t)NNODES * HCH * 4);
  unsigned short* wallb = (unsigned short*)carve((size_t)NCOLS * HCH * 2);
  float*          ball  = (float*)carve((size_t)NCOLS * 4);
  float*          yall  = (float*)carve((size_t)NNODES * NCOLS * 4);
  float*          alpha = (float*)carve((size_t)NEDGES * NHEAD * 4);
  unsigned*       amaxE = (unsigned*)carve((size_t)NNODES * NHEAD * 4);
  float*          denom = (float*)carve((size_t)NNODES * NHEAD * 4);
  float*          agg   = (float*)carve((size_t)NNODES * HCH * 4);

  prep_kernel<<<(NNODES * 128 + 255) / 256, 256, 0, stream>>>(x_orig, mask, fill, xb);

  for (int l = 0; l < 3; ++l) {
    const int Kdim = (l == 0) ? 128 : 256;
    const int base = 4 + 11 * l;
    const float* Wq  = (const float*)d_in[base + 0];
    const float* bq  = (const float*)d_in[base + 1];
    const float* Wk  = (const float*)d_in[base + 2];
    const float* bk  = (const float*)d_in[base + 3];
    const float* Wv  = (const float*)d_in[base + 4];
    const float* bv  = (const float*)d_in[base + 5];
    const float* Ws  = (const float*)d_in[base + 6];
    const float* bs  = (const float*)d_in[base + 7];
    const float* Wb  = (const float*)d_in[base + 8];
    const float* lng = (const float*)d_in[base + 9];
    const float* lnb = (const float*)d_in[base + 10];

    pack_w_kernel<<<(NCOLS * Kdim + 255) / 256, 256, 0, stream>>>(
        Wq, Wk, Wv, Ws, bq, bk, bv, bs, wallb, ball, Kdim);

    const int tiles = ((NNODES + 15) / 16) * (NCOLS / 64);     // 1563 * 16 = 25008
    if (Kdim == 128) {
      gemm_wmma_kernel<128><<<(tiles + 7) / 8, 256, 0, stream>>>(xb, wallb, ball, yall);
    } else {
      gemm_wmma_kernel<256><<<(tiles + 7) / 8, 256, 0, stream>>>(xb, wallb, ball, yall);
    }

    const int zc = NNODES * NHEAD;
    zero_u32_kernel<<<(zc + 255) / 256, 256, 0, stream>>>(amaxE, zc);
    zero_u32_kernel<<<(zc + 255) / 256, 256, 0, stream>>>((unsigned*)denom, zc);
    const int za = NNODES * HCH;
    zero_u32_kernel<<<(za + 255) / 256, 256, 0, stream>>>((unsigned*)agg, za);

    const int eh = NEDGES * NHEAD;
    attn_logits_kernel<<<(eh + 255) / 256, 256, 0, stream>>>(ei, yall, alpha, amaxE);
    attn_accum_kernel<<<(eh + 255) / 256, 256, 0, stream>>>(ei, yall, alpha, amaxE, denom, agg);

    const float*    xres  = (l == 1) ? xfA : nullptr;          // residual only at layer 1
    float*          xout  = (l == 0) ? xfA : (l == 1) ? xfB : (float*)d_out;
    unsigned short* xbout = (l < 2) ? xb : nullptr;
    finish_kernel<<<NNODES, 256, 0, stream>>>(yall, agg, denom, Wb, lng, lnb,
                                              xres, xout, xbout, (l < 2) ? 1 : 0);
  }
}
